// QuantLinear_24885040513624
// MI455X (gfx1250) — matrix operand order repaired and checked by reading, not compiled, not run on hardware
//
#include <hip/hip_runtime.h>

// GPTQ 4-bit fused dequant + GEMM for MI455X (gfx1250, wave32, WMMA).
// y[4, 11008] = x[4, 4096] @ dequant(qweight, qzeros, scales) + bias
// Memory-bound: ~24 MB of weight traffic -> ~1 us at 23.3 TB/s.
//
// Scheme per 256-thread block (8 waves): one 16-column tile, K split 8 ways
// (512 K = exactly 4 GPTQ groups per wave). x is staged once into LDS as f16
// in WMMA-slot order. B fragments are unpacked to EXACT f16 integers (q - z)
// via the 0x6400 magic-number trick (and_or + pk_add_neg only); the group
// scale is folded into the f32 accumulator after each group's 4 chained
// WMMAs (8 FMAs), which is both cheaper and more accurate than scaling B.

#define IN_F      4096
#define OUT_F     11008
#define GROUPSIZE 128
#define TOKENS    4
#define WAVES_PER_BLOCK 8
#define ROW_CHUNKS 513               // 512 chunks of 8 halves + 1 pad chunk per row

typedef _Float16 v16h __attribute__((ext_vector_type(16)));
typedef _Float16 v8h  __attribute__((ext_vector_type(8)));
typedef _Float16 v2h  __attribute__((ext_vector_type(2)));
typedef float    v8f  __attribute__((ext_vector_type(8)));

__global__ __launch_bounds__(256) void gptq_gemm_wmma(
    const float* __restrict__ x,
    const float* __restrict__ scales,
    const float* __restrict__ bias,
    const unsigned* __restrict__ qweight,   // [IN_F/8, OUT_F], 8 K-values / int32
    const unsigned* __restrict__ qzeros,    // [NGROUPS, OUT_F/8], 8 N-values / int32
    float* __restrict__ out)                // [TOKENS, OUT_F]
{
    __shared__ __align__(16) _Float16 xl[TOKENS * ROW_CHUNKS * 8]; // ~32.8 KB
    __shared__ float red[WAVES_PER_BLOCK][TOKENS][16];

    const int lane = threadIdx.x & 31;
    const int wave = threadIdx.x >> 5;        // 0..7 -> K partition
    const int n0   = blockIdx.x * 16;
    const int ncol = n0 + (lane & 15);        // B/D: N = lane % 16
    const int hsel = lane >> 4;               // 0: K 0-15 of tile, 1: K 16-31
    const int m    = lane & 15;

    // ---- Stage x -> LDS as f16, permuted into WMMA slot order per 8-chunk:
    // position p holds x[c + {0,4,1,5,2,6,3,7}[p]] (matches magic-unpack pairs).
    #pragma unroll
    for (int it = 0; it < (TOKENS * IN_F / 8) / 256; ++it) {
        const int ci  = threadIdx.x + it * 256;     // flat 8-chunk index
        const int row = ci >> 9;                    // 512 chunks per token row
        const int cr  = ci & 511;
        const float4* xp = (const float4*)(x + ci * 8);
        const float4 lo4 = xp[0];                   // x[c .. c+3]
        const float4 hi4 = xp[1];                   // x[c+4 .. c+7]
        v8h c;
        c[0] = (_Float16)lo4.x; c[1] = (_Float16)hi4.x;
        c[2] = (_Float16)lo4.y; c[3] = (_Float16)hi4.y;
        c[4] = (_Float16)lo4.z; c[5] = (_Float16)hi4.z;
        c[6] = (_Float16)lo4.w; c[7] = (_Float16)hi4.w;
        ((v8h*)xl)[row * ROW_CHUNKS + cr] = c;
    }
    __syncthreads();

    const v8h* xv = (const v8h*)xl;
    const int arow = (m & (TOKENS - 1)) * ROW_CHUNKS;  // rows 4-15 duplicate 0-3
                                                       // (their D rows are discarded)
    v8f acc = {};

    #pragma unroll
    for (int gi = 0; gi < 4; ++gi) {
        const int g = wave * 4 + gi;          // GPTQ group (128 K values)

        const unsigned zq =
            (qzeros[g * (OUT_F / 8) + (ncol >> 3)] >> ((ncol & 7) * 4)) & 0xFu;
        const float s = scales[g * OUT_F + ncol];
        // f16 bits of (1024 + z), z = zq + 1: exact, built bitwise (no cvt).
        const unsigned zb = 0x6401u + zq;
        const v2h z2 = __builtin_bit_cast(v2h, (zb << 16) | zb);

        v8f tmp = {};                         // unscaled partial tile for this group
        #pragma unroll
        for (int kt = 0; kt < 4; ++kt) {
            const int k0  = g * GROUPSIZE + kt * 32;
            const int row = (k0 >> 3) + hsel * 2;
            const unsigned w0 = qweight[row * OUT_F + ncol];
            const unsigned w1 = qweight[(row + 1) * OUT_F + ncol];

            // B fragment: exact f16 integers (q - z) in slot order.
            v16h b;
            #pragma unroll
            for (int j = 0; j < 4; ++j) {
                const unsigned t0 = ((w0 >> (4 * j)) & 0x000F000Fu) | 0x64006400u;
                const unsigned t1 = ((w1 >> (4 * j)) & 0x000F000Fu) | 0x64006400u;
                const v2h h0 = __builtin_bit_cast(v2h, t0) - z2;   // exact
                const v2h h1 = __builtin_bit_cast(v2h, t1) - z2;   // exact
                b[2 * j]     = h0[0]; b[2 * j + 1] = h0[1];
                b[8 + 2 * j] = h1[0]; b[9 + 2 * j] = h1[1];
            }

            // A fragment: two 16B LDS chunks (already f16, already permuted).
            const int cidx = arow + (k0 >> 3) + hsel * 2;
            const v8h alo = xv[cidx];
            const v8h ahi = xv[cidx + 1];
            const v16h a = __builtin_shufflevector(
                alo, ahi, 0, 1, 2, 3, 4, 5, 6, 7, 8, 9, 10, 11, 12, 13, 14, 15);

            tmp = __builtin_amdgcn_wmma_f32_16x16x32_f16(
                false, a, false, b, (short)0, tmp, false, false);
        }

        // Fold the per-(group, column) scale in f32: s is per-lane on the D tile.
        #pragma unroll
        for (int i = 0; i < 8; ++i) acc[i] = fmaf(s, tmp[i], acc[i]);
    }

    // ---- Reduce the 8 K-partitions across waves via LDS, add bias, store.
    if (lane < 16) {
        #pragma unroll
        for (int r = 0; r < TOKENS; ++r)      // D row r -> acc[r] in lanes 0-15
            red[wave][r][lane] = acc[r];
    }
    __syncthreads();

    if (threadIdx.x < TOKENS * 16) {
        const int mo = threadIdx.x >> 4;
        const int nn = threadIdx.x & 15;
        float sum = bias[n0 + nn];
        #pragma unroll
        for (int w = 0; w < WAVES_PER_BLOCK; ++w)
            sum += red[w][mo][nn];
        out[mo * OUT_F + n0 + nn] = sum;
    }
}

extern "C" void kernel_launch(void* const* d_in, const int* in_sizes, int n_in,
                              void* d_out, int out_size, void* d_ws, size_t ws_size,
                              hipStream_t stream) {
    const float*    x       = (const float*)d_in[0];
    const float*    scales  = (const float*)d_in[1];
    const float*    bias    = (const float*)d_in[2];
    const unsigned* qweight = (const unsigned*)d_in[3];
    const unsigned* qzeros  = (const unsigned*)d_in[4];
    float*          out     = (float*)d_out;

    dim3 grid(OUT_F / 16);   // 688 column tiles
    dim3 block(256);         // 8 waves, K split 8 ways
    hipLaunchKernelGGL(gptq_gemm_wmma, grid, block, 0, stream,
                       x, scales, bias, qweight, qzeros, out);
}
